// LocalFeatureTransformer_11261404250305
// MI455X (gfx1250) — matrix operand order repaired and checked by reading, not compiled, not run on hardware
//
#include <hip/hip_runtime.h>
#include <hip/hip_bf16.h>
#include <stdint.h>

typedef __attribute__((ext_vector_type(16))) _Float16 v16h;
typedef __attribute__((ext_vector_type(8)))  _Float16 v8h;
typedef __attribute__((ext_vector_type(4)))  _Float16 v4h;
typedef __attribute__((ext_vector_type(8)))  float    v8f;
typedef __attribute__((ext_vector_type(4)))  float    fvec4;
typedef int int4_vs __attribute__((vector_size(16)));   // matches builtin prototype

#define C_DIM   256
#define NHEAD   8
#define L_SEQ   4800
#define A_AG    128
#define EPS_LA  1e-6f
#define EPS_LN  1e-5f
#define LDB     40   // column stride (halves) of panelized f16 weights: 80B, 16B-aligned
#define PANEL_H (256 * LDB)          // halves per (32k x 256col) panel

// ---- async global->LDS copy (CDNA5 ASYNCcnt path) with safe fallback ----
#if defined(__has_builtin)
#if __has_builtin(__builtin_amdgcn_global_load_async_to_lds_b128) && \
    __has_builtin(__builtin_amdgcn_s_wait_asynccnt)
#define USE_ASYNC_LDS 1
#endif
#endif

__device__ __forceinline__ void lds_copy_b128(v8h* dst_lds, const v8h* src_glb) {
#ifdef USE_ASYNC_LDS
  __builtin_amdgcn_global_load_async_to_lds_b128(
      (__attribute__((address_space(1))) int4_vs*)src_glb,
      (__attribute__((address_space(3))) int4_vs*)dst_lds, 0, 0);
#else
  *dst_lds = *src_glb;
#endif
}
__device__ __forceinline__ void lds_copy_fence() {
#ifdef USE_ASYNC_LDS
  __builtin_amdgcn_s_wait_asynccnt(0);
#endif
}
// flat 20KB panel fill: 1280 x b128 over 256 threads
__device__ __forceinline__ void fill_panel(_Float16* dst, const _Float16* src, int t) {
#pragma unroll
  for (int i = 0; i < 5; ++i)
    lds_copy_b128((v8h*)dst + t + i * 256, (const v8h*)src + t + i * 256);
}

__device__ __forceinline__ float elu1(float x) { return x > 0.f ? x + 1.f : __expf(x); }

// ---------------- WMMA fragment helpers (v_wmma_f32_16x16x32_f16) ----------------
// A 16x32 f16 (ISA 7.12.2): lane (m,g) holds two contiguous 8-half runs -> 2 x b128.
__device__ __forceinline__ v16h frag_a(const _Float16* p, int ld) {
  const int m = threadIdx.x & 15, g = (threadIdx.x >> 4) & 1;
  const _Float16* q = p + m * ld + g * 8;
  v8h lo = *(const v8h*)(q);
  v8h hi = *(const v8h*)(q + 16);
  return __builtin_shufflevector(lo, hi, 0, 1, 2, 3, 4, 5, 6, 7,
                                 8, 9, 10, 11, 12, 13, 14, 15);
}
// B 32x16 f16 from COLUMN-major LDS panel (p[col*ld + k]) -> 2 x b128.
__device__ __forceinline__ v16h frag_b_cm(const _Float16* p, int ld) {
  const int n = threadIdx.x & 15, g = (threadIdx.x >> 4) & 1;
  const _Float16* q = p + n * ld + g * 8;
  v8h lo = *(const v8h*)(q);
  v8h hi = *(const v8h*)(q + 16);
  return __builtin_shufflevector(lo, hi, 0, 1, 2, 3, 4, 5, 6, 7,
                                 8, 9, 10, 11, 12, 13, 14, 15);
}
__device__ __forceinline__ v8f wmma32(v16h a, v16h b, v8f c) {
  return __builtin_amdgcn_wmma_f32_16x16x32_f16(false, a, false, b, (short)0, c, false, false);
}

// ---------------- utility ----------------
__global__ void zero_kernel(float* __restrict__ p, int n) {
  int i = blockIdx.x * blockDim.x + threadIdx.x;
  if (i < n) p[i] = 0.f;
}

// ---------------- weight pre-panelization: fp32 (K x N) -> f16 LDS-image -----------
// out[(panel*ncb + cb)*PANEL_H + col*LDB + k]   (grid: K/32 x N/256)
__global__ void __launch_bounds__(256)
panelize_kernel(const float* __restrict__ W, int N, _Float16* __restrict__ out) {
  const int t = threadIdx.x;
  const int k0 = blockIdx.x * 32;
  const int cb = blockIdx.y, ncb = gridDim.y;
  _Float16* op = out + ((size_t)(blockIdx.x * ncb + cb)) * PANEL_H;
#pragma unroll
  for (int u = 0; u < 8; ++u) {
    int q = t + u * 256;                 // 2048 float4 chunks
    int r = q >> 6, c4 = (q & 63) << 2;
    fvec4 x = *(const fvec4*)&W[(size_t)(k0 + r) * N + cb * 256 + c4];
#pragma unroll
    for (int j = 0; j < 4; ++j) op[(c4 + j) * LDB + r] = (_Float16)x[j];
  }
}

// ---------------- Stage 1: fused K/V projection + encoder linear-attn stats ----------
__global__ void __launch_bounds__(256)
enc_kv_kernel(const float* __restrict__ feat0, const float* __restrict__ feat1,
              const _Float16* __restrict__ Wkf, const _Float16* __restrict__ Wvf,
              float* __restrict__ KV, float* __restrict__ Ksum) {
  __shared__ alignas(16) char smem[43008];
  _Float16* Xp  = (_Float16*)smem;              // 32x32 row-major      (2048 B)
  _Float16* Wkt = (_Float16*)(smem + 2048);     // panel               (20480 B)
  _Float16* Wvt = (_Float16*)(smem + 22528);    // panel               (20480 B)
  _Float16* Kst = (_Float16*)smem;              // overlay col-major [d][s]
  _Float16* Vst = (_Float16*)(smem + 20480);    // overlay col-major [v][s]

  const int t  = threadIdx.x;
  const int wv = t >> 5;
  const int n  = blockIdx.y;
  const int s0 = blockIdx.x * 32;
  const float* feat = (n < 4) ? (feat0 + (size_t)n * L_SEQ * C_DIM)
                              : (feat1 + (size_t)(n - 4) * L_SEQ * C_DIM);
  const int mt = wv & 1;
  const int nbase = (wv >> 1) * 64;
  v8f accK[4] = {}; v8f accV[4] = {};

  for (int k0 = 0; k0 < C_DIM; k0 += 32) {
    {  // X panel: one float4 per thread
      int r = t >> 3, c4 = (t & 7) << 2;
      fvec4 x = *(const fvec4*)&feat[(size_t)(s0 + r) * C_DIM + k0 + c4];
      v4h h;
#pragma unroll
      for (int j = 0; j < 4; ++j) h[j] = (_Float16)x[j];
      *(v4h*)&Xp[r * 32 + c4] = h;
    }
    fill_panel(Wkt, Wkf + (size_t)(k0 >> 5) * PANEL_H, t);
    fill_panel(Wvt, Wvf + (size_t)(k0 >> 5) * PANEL_H, t);
    lds_copy_fence();
    __syncthreads();
    v16h a = frag_a(Xp + mt * 16 * 32, 32);
    v16h bk[4], bv[4];
#pragma unroll
    for (int j = 0; j < 4; ++j) {
      bk[j] = frag_b_cm(Wkt + (nbase + j * 16) * LDB, LDB);
      bv[j] = frag_b_cm(Wvt + (nbase + j * 16) * LDB, LDB);
    }
#pragma unroll
    for (int j = 0; j < 4; ++j) {
      accK[j] = wmma32(a, bk[j], accK[j]);
      accV[j] = wmma32(a, bv[j], accV[j]);
    }
    __syncthreads();
  }
  // epilogue into overlaid LDS, column-major; per-lane tile rows contiguous -> b128 stores
  const int nn = t & 15, g = (t >> 4) & 1;
#pragma unroll
  for (int j = 0; j < 4; ++j) {
    v8h kc, vc;
#pragma unroll
    for (int r = 0; r < 8; ++r) {
      kc[r] = (_Float16)elu1(accK[j][r]);
      vc[r] = (_Float16)(accV[j][r] * (1.0f / (float)L_SEQ));
    }
    int col = nbase + j * 16 + nn;
    *(v8h*)(Kst + col * LDB + mt * 16 + g * 8) = kc;
    *(v8h*)(Vst + col * LDB + mt * 16 + g * 8) = vc;
  }
  __syncthreads();
  {  // Ksum: one column per thread (contiguous)
    float s = 0.f;
#pragma unroll
    for (int r = 0; r < 32; ++r) s += (float)Kst[t * LDB + r];
    atomicAdd(&Ksum[n * C_DIM + t], s);
  }
  {  // per-head KV partial: wave == head; K^T is plain frag_a on col-major Kst
    const int h = wv;
#pragma unroll
    for (int mt2 = 0; mt2 < 2; ++mt2) {
      v16h a = frag_a(Kst + (h * 32 + mt2 * 16) * LDB, LDB);
#pragma unroll
      for (int nt2 = 0; nt2 < 2; ++nt2) {
        v16h b = frag_b_cm(Vst + (h * 32 + nt2 * 16) * LDB, LDB);
        v8f c = {};
        c = wmma32(a, b, c);
#pragma unroll
        for (int r = 0; r < 8; ++r) {
          int dr = mt2 * 16 + r + 8 * g;
          int dc = nt2 * 16 + nn;
          atomicAdd(&KV[(((size_t)n * NHEAD + h) * 32 + dr) * 32 + dc], c[r]);
        }
      }
    }
  }
}

// ---------------- generic 32-row GEMM, activation epilogue -------------------------
// W pre-panelized with ncb col-blocks; this block handles col-block blockIdx.y.
__global__ void __launch_bounds__(256)
gemm_act_kernel(const float* __restrict__ X, int ldx, int kdim,
                const _Float16* __restrict__ Wf,
                float* __restrict__ Y, int ldy, int act) {
  __shared__ alignas(16) _Float16 Xp[32 * 32];
  __shared__ alignas(16) _Float16 Wt[PANEL_H];
  const int t = threadIdx.x, wv = t >> 5;
  const int rowblk = blockIdx.x * 32;
  const int cb = blockIdx.y, ncb = gridDim.y;
  const int mt = wv & 1, nbase = (wv >> 1) * 64;
  v8f acc[4] = {};
  for (int k0 = 0; k0 < kdim; k0 += 32) {
    {
      int r = t >> 3, c4 = (t & 7) << 2;
      fvec4 x = *(const fvec4*)&X[(size_t)(rowblk + r) * ldx + k0 + c4];
      v4h h;
#pragma unroll
      for (int j = 0; j < 4; ++j) h[j] = (_Float16)x[j];
      *(v4h*)&Xp[r * 32 + c4] = h;
    }
    fill_panel(Wt, Wf + (size_t)((k0 >> 5) * ncb + cb) * PANEL_H, t);
    lds_copy_fence();
    __syncthreads();
    v16h a = frag_a(Xp + mt * 16 * 32, 32);
    v16h b[4];
#pragma unroll
    for (int j = 0; j < 4; ++j) b[j] = frag_b_cm(Wt + (nbase + j * 16) * LDB, LDB);
#pragma unroll
    for (int j = 0; j < 4; ++j) acc[j] = wmma32(a, b[j], acc[j]);
    __syncthreads();
  }
  const int nn = t & 15, g = (t >> 4) & 1;
#pragma unroll
  for (int j = 0; j < 4; ++j)
#pragma unroll
    for (int r = 0; r < 8; ++r) {
      float v = acc[j][r];
      if (act == 1)      v = elu1(v);
      else if (act == 2) v = fmaxf(v, 0.f);
      Y[(size_t)(rowblk + mt * 16 + r + 8 * g) * ldy + cb * 256 + nbase + j * 16 + nn] = v;
    }
}

// ---------------- 32-row GEMM + LayerNorm epilogue (+ optional concat fill) --------
__global__ void __launch_bounds__(256)
gemm_ln_kernel(const float* __restrict__ X, int ldx, int kdim,
               const _Float16* __restrict__ Wf,
               const float* __restrict__ gamma, const float* __restrict__ beta,
               float* __restrict__ Y, int ldy, int ycoloff,
               const float* __restrict__ agents_fill) {
  __shared__ alignas(16) _Float16 Xp[32 * 32];
  __shared__ alignas(16) _Float16 Wt[PANEL_H];
  __shared__ float Ys[32 * 256];
  __shared__ float mu[32], rs[32];
  const int t = threadIdx.x, wv = t >> 5;
  const int rowblk = blockIdx.x * 32;
  const int mt = wv & 1, nbase = (wv >> 1) * 64;
  v8f acc[4] = {};
  for (int k0 = 0; k0 < kdim; k0 += 32) {
    {
      int r = t >> 3, c4 = (t & 7) << 2;
      fvec4 x = *(const fvec4*)&X[(size_t)(rowblk + r) * ldx + k0 + c4];
      v4h h;
#pragma unroll
      for (int j = 0; j < 4; ++j) h[j] = (_Float16)x[j];
      *(v4h*)&Xp[r * 32 + c4] = h;
    }
    fill_panel(Wt, Wf + (size_t)(k0 >> 5) * PANEL_H, t);
    lds_copy_fence();
    __syncthreads();
    v16h a = frag_a(Xp + mt * 16 * 32, 32);
    v16h b[4];
#pragma unroll
    for (int j = 0; j < 4; ++j) b[j] = frag_b_cm(Wt + (nbase + j * 16) * LDB, LDB);
#pragma unroll
    for (int j = 0; j < 4; ++j) acc[j] = wmma32(a, b[j], acc[j]);
    __syncthreads();
  }
  const int nn = t & 15, g = (t >> 4) & 1;
#pragma unroll
  for (int j = 0; j < 4; ++j)
#pragma unroll
    for (int r = 0; r < 8; ++r)
      Ys[(mt * 16 + r + 8 * g) * 256 + nbase + j * 16 + nn] = acc[j][r];
  __syncthreads();
  if (t < 32) {
    float m = 0.f;
    for (int c = 0; c < 256; ++c) m += Ys[t * 256 + c];
    m *= (1.f / 256.f);
    float v = 0.f;
    for (int c = 0; c < 256; ++c) { float d = Ys[t * 256 + c] - m; v += d * d; }
    v *= (1.f / 256.f);
    mu[t] = m;
    rs[t] = rsqrtf(v + EPS_LN);
  }
  __syncthreads();
#pragma unroll
  for (int u = 0; u < 8; ++u) {          // 2048 float4 chunks
    int q = t + u * 256;
    int r = q >> 6, c4 = (q & 63) << 2;
    int grow = rowblk + r;
    fvec4 y = *(const fvec4*)&Ys[r * 256 + c4];
    fvec4 gm = *(const fvec4*)&gamma[c4];
    fvec4 bt = *(const fvec4*)&beta[c4];
    fvec4 o;
#pragma unroll
    for (int j = 0; j < 4; ++j) o[j] = (y[j] - mu[r]) * rs[r] * gm[j] + bt[j];
    *(fvec4*)&Y[(size_t)grow * ldy + ycoloff + c4] = o;
    if (agents_fill)
      *(fvec4*)&Y[(size_t)grow * ldy + c4] =
          *(const fvec4*)&agents_fill[(grow & (A_AG - 1)) * C_DIM + c4];
  }
}

// ---------------- agent attention apply (tiny; VALU) -------------------------------
__global__ void __launch_bounds__(256)
attn_apply_kernel(const float* __restrict__ Qa, const float* __restrict__ KV,
                  const float* __restrict__ Ksum, float* __restrict__ Out) {
  __shared__ float qrow[256];
  __shared__ float zs[8];
  const int t = threadIdx.x;
  const int n = blockIdx.x >> 7;
  const int l = blockIdx.x & 127;
  qrow[t] = Qa[l * C_DIM + t];
  __syncthreads();
  if (t < 8) {
    float d = 0.f;
#pragma unroll
    for (int k = 0; k < 32; ++k) d += qrow[t * 32 + k] * Ksum[n * C_DIM + t * 32 + k];
    zs[t] = (float)L_SEQ / (d + EPS_LA);
  }
  __syncthreads();
  const int h = t >> 5, dv = t & 31;
  float s = 0.f;
#pragma unroll 8
  for (int d = 0; d < 32; ++d)
    s += qrow[h * 32 + d] * KV[(((size_t)n * NHEAD + h) * 32 + d) * 32 + dv];
  Out[((size_t)n * A_AG + l) * C_DIM + t] = s * zs[h];
}

// ---------------- full-attention stats: KV2f16 = panelized(elu1(u)^T @ (u/128)) ----
__global__ void __launch_bounds__(256)
ustats_kernel(const float* __restrict__ U, _Float16* __restrict__ KV2f) {
  __shared__ alignas(16) _Float16 Kut[256 * LDB];   // col-major: [d col][s row]
  __shared__ alignas(16) _Float16 Vut[256 * LDB];   // col-major: [v col][s row]
  const int t = threadIdx.x, wv = t >> 5;
  const int mb = blockIdx.x;      // d-panel (8 x 32)
  const int n  = blockIdx.y;
  const int mt = wv & 1, nbase = (wv >> 1) * 64;
  v8f acc[4] = {};
  for (int s0 = 0; s0 < A_AG; s0 += 32) {
#pragma unroll
    for (int u = 0; u < 8; ++u) {        // 2048 float4 chunks
      int q = t + u * 256;
      int r = q >> 6, c4 = (q & 63) << 2;
      fvec4 x = *(const fvec4*)&U[((size_t)n * A_AG + s0 + r) * C_DIM + c4];
#pragma unroll
      for (int j = 0; j < 4; ++j) {
        Kut[(c4 + j) * LDB + r] = (_Float16)elu1(x[j]);
        Vut[(c4 + j) * LDB + r] = (_Float16)(x[j] * (1.f / (float)A_AG));
      }
    }
    __syncthreads();
    v16h a = frag_a(Kut + (mb * 32 + mt * 16) * LDB, LDB);   // K^T via col-major
    v16h b[4];
#pragma unroll
    for (int j = 0; j < 4; ++j) b[j] = frag_b_cm(Vut + (nbase + j * 16) * LDB, LDB);
#pragma unroll
    for (int j = 0; j < 4; ++j) acc[j] = wmma32(a, b[j], acc[j]);
    __syncthreads();
  }
  // panelized f16 store: per tile column, 8 contiguous k's -> one b128
  const int nn = t & 15, g = (t >> 4) & 1;
#pragma unroll
  for (int j = 0; j < 4; ++j) {
    v8h h;
#pragma unroll
    for (int r = 0; r < 8; ++r) h[r] = (_Float16)acc[j][r];
    int dc = nbase + j * 16 + nn;
    *(v8h*)(KV2f + ((size_t)(n * 8 + mb) * 256 + dc) * LDB + mt * 16 + g * 8) = h;
  }
}

__global__ void ksum2_kernel(const float* __restrict__ U, float* __restrict__ Ksum2) {
  const int n = blockIdx.x, t = threadIdx.x;
  float s = 0.f;
  for (int i = 0; i < A_AG; ++i) s += elu1(U[((size_t)n * A_AG + i) * C_DIM + t]);
  Ksum2[n * C_DIM + t] = s;
}

// ---------------- final: out = feat + (elu1(feat) @ KV2) * Z * 128 -----------------
__global__ void __launch_bounds__(256)
final_kernel(const float* __restrict__ feat0, const float* __restrict__ feat1,
             const _Float16* __restrict__ KV2f, const float* __restrict__ Ksum2,
             float* __restrict__ out) {
  __shared__ alignas(16) _Float16 Qp[32 * 32];
  __shared__ alignas(16) _Float16 Wt[PANEL_H];
  __shared__ float zacc[32];
  const int t = threadIdx.x, wv = t >> 5;
  const int n  = blockIdx.y;
  const int s0 = blockIdx.x * 32;
  const float* feat = (n < 4) ? (feat0 + (size_t)n * L_SEQ * C_DIM)
                              : (feat1 + (size_t)(n - 4) * L_SEQ * C_DIM);
  float* outp = out + (size_t)n * L_SEQ * C_DIM;  // out = concat(out0, out1)
  if (t < 32) zacc[t] = 0.f;
  const int mt = wv & 1, nbase = (wv >> 1) * 64;
  v8f acc[4] = {};
  for (int k0 = 0; k0 < C_DIM; k0 += 32) {
    {
      int r = t >> 3, c4 = (t & 7) << 2;
      fvec4 x = *(const fvec4*)&feat[(size_t)(s0 + r) * C_DIM + k0 + c4];
      v4h h;
#pragma unroll
      for (int j = 0; j < 4; ++j) h[j] = (_Float16)elu1(x[j]);
      *(v4h*)&Qp[r * 32 + c4] = h;
    }
    fill_panel(Wt, KV2f + (size_t)(n * 8 + (k0 >> 5)) * PANEL_H, t);
    lds_copy_fence();
    __syncthreads();
    if (t < 32) {
      float part = 0.f;
#pragma unroll
      for (int kk = 0; kk < 32; ++kk)
        part += (float)Qp[t * 32 + kk] * Ksum2[n * C_DIM + k0 + kk];
      zacc[t] += part;
    }
    v16h a = frag_a(Qp + mt * 16 * 32, 32);
    v16h b[4];
#pragma unroll
    for (int j = 0; j < 4; ++j) b[j] = frag_b_cm(Wt + (nbase + j * 16) * LDB, LDB);
#pragma unroll
    for (int j = 0; j < 4; ++j) acc[j] = wmma32(a, b[j], acc[j]);
    __syncthreads();
  }
  const int nn = t & 15, g = (t >> 4) & 1;
#pragma unroll
  for (int j = 0; j < 4; ++j)
#pragma unroll
    for (int r = 0; r < 8; ++r) {
      int lr = mt * 16 + r + 8 * g;
      int col = nbase + j * 16 + nn;
      float scale = (float)A_AG / (zacc[lr] + EPS_LA);
      size_t idx = (size_t)(s0 + lr) * C_DIM + col;
      outp[idx] = feat[idx] + acc[j][r] * scale;
    }
}

// ---------------- host launcher ----------------------------------------------------
extern "C" void kernel_launch(void* const* d_in, const int* in_sizes, int n_in,
                              void* d_out, int out_size, void* d_ws, size_t ws_size,
                              hipStream_t stream) {
  (void)in_sizes; (void)n_in; (void)out_size; (void)ws_size;
  const float* agents = (const float*)d_in[0];
  const float* feat0  = (const float*)d_in[1];
  const float* feat1  = (const float*)d_in[2];
  const float* Wq     = (const float*)d_in[3];
  const float* Wk     = (const float*)d_in[4];
  const float* Wv     = (const float*)d_in[5];
  const float* Wm     = (const float*)d_in[6];
  const float* W1     = (const float*)d_in[7];
  const float* W2     = (const float*)d_in[8];
  const float* g1     = (const float*)d_in[9];
  const float* b1     = (const float*)d_in[10];
  const float* g2     = (const float*)d_in[11];
  const float* b2     = (const float*)d_in[12];
  float* out = (float*)d_out;
  float* ws  = (float*)d_ws;

  // workspace layout: fp32 regions
  float* KV    = ws;             //  8*8*32*32 = 65536
  float* Ksum  = ws + 65536;     //  8*256     = 2048
  float* Qa    = ws + 67584;     //  128*256   = 32768
  float* Msg   = ws + 100352;    //  8*128*256 = 262144
  float* Cat   = ws + 362496;    //  8*128*512 = 524288
  float* H2    = ws + 886784;    //  8*128*512 = 524288
  float* U     = ws + 1411072;   //  8*128*256 = 262144
  float* Ksum2 = ws + 1673216;   //  8*256     = 2048
  // f16 regions (panelized weights + KV2), 16B-aligned
  _Float16* hb   = (_Float16*)(ws + 1675264);
  _Float16* KV2f = hb;               // 8*8 panels  = 655360 halves
  _Float16* Wqf  = hb + 655360;      // 8 panels    =  81920
  _Float16* Wkf  = hb + 737280;      // 8 panels
  _Float16* Wvf  = hb + 819200;      // 8 panels
  _Float16* Wmf  = hb + 901120;      // 8 panels
  _Float16* W1f  = hb + 983040;      // 16 panels x 2 cb = 327680
  _Float16* W2f  = hb + 1310720;     // 16 panels x 1 cb = 163840

  // one-time (per launch) weight panelization; all independent
  panelize_kernel<<<dim3(8, 1), 256, 0, stream>>>(Wq, 256, Wqf);
  panelize_kernel<<<dim3(8, 1), 256, 0, stream>>>(Wk, 256, Wkf);
  panelize_kernel<<<dim3(8, 1), 256, 0, stream>>>(Wv, 256, Wvf);
  panelize_kernel<<<dim3(8, 1), 256, 0, stream>>>(Wm, 256, Wmf);
  panelize_kernel<<<dim3(16, 2), 256, 0, stream>>>(W1, 512, W1f);
  panelize_kernel<<<dim3(16, 1), 256, 0, stream>>>(W2, 256, W2f);
  zero_kernel<<<264, 256, 0, stream>>>(ws, 67584);  // KV + Ksum accumulators

  enc_kv_kernel<<<dim3(L_SEQ / 32, 8), 256, 0, stream>>>(feat0, feat1, Wkf, Wvf, KV, Ksum);
  // Qa = elu1(agents @ Wq)  (shared across batch)
  gemm_act_kernel<<<dim3(A_AG / 32, 1), 256, 0, stream>>>(agents, 256, 256, Wqf, Qa, 256, 1);
  attn_apply_kernel<<<8 * A_AG, 256, 0, stream>>>(Qa, KV, Ksum, Msg);
  // Cat[:, 256:512] = LN(Msg @ Wm);  Cat[:, 0:256] = agents
  gemm_ln_kernel<<<32, 256, 0, stream>>>(Msg, 256, 256, Wmf, g1, b1, Cat, 512, 256, agents);
  // H2 = relu(Cat @ W1)
  gemm_act_kernel<<<dim3(32, 2), 256, 0, stream>>>(Cat, 512, 512, W1f, H2, 512, 2);
  // U = LN(H2 @ W2)
  gemm_ln_kernel<<<32, 256, 0, stream>>>(H2, 512, 512, W2f, g2, b2, U, 256, 0, nullptr);
  ustats_kernel<<<dim3(8, 8), 256, 0, stream>>>(U, KV2f);
  ksum2_kernel<<<8, 256, 0, stream>>>(U, Ksum2);
  final_kernel<<<dim3(L_SEQ / 32, 8), 256, 0, stream>>>(feat0, feat1, KV2f, Ksum2, out);
}
